// ConvLSTM_41369124995471
// MI455X (gfx1250) — matrix operand order repaired and checked by reading, not compiled, not run on hardware
//
#include <hip/hip_runtime.h>

// ---------------------------------------------------------------------------
// CDNA5 (gfx1250, wave32) ConvLSTM: implicit-GEMM 3x3 convs on
// v_wmma_f32_16x16x32_bf16, async global->LDS staging (ASYNCcnt path),
// bf16 NHWC intermediates, fp32 accumulate.
// ---------------------------------------------------------------------------

typedef __bf16 bf16;
typedef __attribute__((ext_vector_type(16))) __bf16 v16bf;
typedef __attribute__((ext_vector_type(8)))  __bf16 v8bf;
typedef __attribute__((ext_vector_type(8)))  float   v8f;
typedef __attribute__((ext_vector_type(4)))  unsigned int u32x4;
typedef int v4i_ __attribute__((vector_size(4 * sizeof(int))));  // matches builtin param

#define CH    64          // Cout == Cin for all WMMA convs
#define KTOT  576         // 9 taps * 64 cin, k = (ky*3+kx)*64 + cin
#define NTHR  384         // 12 wave32s: 3 gates x 4 n-tiles

#if defined(__gfx1250__) && __has_builtin(__builtin_amdgcn_global_load_async_to_lds_b128)
#define HAVE_ASYNC_LDS 1
#endif

__device__ __forceinline__ float sigmoidf_(float x) { return 1.0f / (1.0f + __expf(-x)); }

__device__ __forceinline__ void wait_async_then_ready()
{
#if defined(HAVE_ASYNC_LDS)
#if __has_builtin(__builtin_amdgcn_s_wait_asynccnt)
  __builtin_amdgcn_s_wait_asynccnt(0);
#else
  asm volatile("s_wait_asynccnt 0x0" ::: "memory");
#endif
#endif
}

// Stage a 3-row x 18-col x 64-ch halo patch (bf16 NHWC source) into LDS,
// zero-filling out-of-bounds pixels. 54 pixels * 8 chunks of 16B.
// In-bounds chunks go through the gfx1250 async global->LDS DMA path
// (no VGPR round-trip, tracked on ASYNCcnt); OOB chunks are ds-stored zeros.
__device__ __forceinline__ void stage_patch(bf16* lds, const bf16* __restrict__ src,
                                            int H, int W, int oy, int ox0, int tid)
{
  for (int idx = tid; idx < 54 * 8; idx += NTHR) {
    int pix = idx >> 3, c8 = idx & 7;
    int row = pix / 18, col = pix - row * 18;
    int y = oy + row - 1, x = ox0 + col - 1;
    bool inb = (y >= 0 && y < H && x >= 0 && x < W);
#if defined(HAVE_ASYNC_LDS)
    if (inb) {
      __attribute__((address_space(1))) v4i_* gp =
          (__attribute__((address_space(1))) v4i_*)(src + ((size_t)(y * W + x) * CH + c8 * 8));
      __attribute__((address_space(3))) v4i_* lp =
          (__attribute__((address_space(3))) v4i_*)(lds + pix * CH + c8 * 8);
      __builtin_amdgcn_global_load_async_to_lds_b128(gp, lp, 0, 0);
    } else {
      u32x4 z = {0u, 0u, 0u, 0u};
      *(u32x4*)(lds + pix * CH + c8 * 8) = z;
    }
#else
    u32x4 v = {0u, 0u, 0u, 0u};
    if (inb)
      v = *(const u32x4*)(src + ((size_t)(y * W + x) * CH + c8 * 8));
    *(u32x4*)(lds + pix * CH + c8 * 8) = v;
#endif
  }
  wait_async_then_ready();   // this wave's async LDS writes visible; barrier follows
}

// One 16(M=pixels) x 16(N=couts) tile of a 3x3/Cin=64 conv: 18 WMMA K-steps.
// A fragment per ISA 7.12.2 (16-bit A 16x32): lane<16 elems i -> k = i(+8 if i>=8),
// lane>=16 -> +8 more. With our K-order each lane-step is one pixel + 32 chans.
// B fragment: same k order as A: lane reads kbase+{0..7},{16..23} (+8 for hi lanes).
__device__ __forceinline__ v8f conv_tile_576(const bf16* __restrict__ wrow, // Wb + n*KTOT
                                             const bf16* lds_patch, int lane)
{
  v8f acc;
#pragma unroll
  for (int r = 0; r < 8; ++r) acc[r] = 0.0f;
  const int hi = (lane >> 4) & 1;
  const int m  = lane & 15;
#pragma unroll
  for (int s = 0; s < 18; ++s) {
    const int r  = s >> 1;           // tap 0..8
    const int cb = (s & 1) << 5;     // cin base 0 / 32
    const int ky = r / 3;
    const int kx = r - ky * 3;
    const bf16* ap = lds_patch + ((ky * 18 + (m + kx)) * CH + cb + hi * 8);
    v8bf a0 = *(const v8bf*)(ap);        // cins +{0..7}  (hi: +{8..15})
    v8bf a1 = *(const v8bf*)(ap + 16);   // cins +{16..23}(hi: +{24..31})
    const bf16* bp = wrow + (s << 5) + hi * 8;
    if (s + 2 < 18) __builtin_prefetch(wrow + ((s + 2) << 5), 0, 1);
    v8bf b0 = *(const v8bf*)(bp);
    v8bf b1 = *(const v8bf*)(bp + 16);
    v16bf A = __builtin_shufflevector(a0, a1, 0,1,2,3,4,5,6,7,8,9,10,11,12,13,14,15);
    v16bf B = __builtin_shufflevector(b0, b1, 0,1,2,3,4,5,6,7,8,9,10,11,12,13,14,15);
    acc = __builtin_amdgcn_wmma_f32_16x16x32_bf16(false, A, false, B, (short)0, acc,
                                                  false, false);
  }
  return acc;
}

// ---------------------------------------------------------------------------
// Convert fp32 weights [cout][cin=64][3][3] -> bf16 [cout][(ky*3+kx)*64+cin]
// ---------------------------------------------------------------------------
__global__ void wconv_kernel(const float* __restrict__ src, bf16* __restrict__ dst)
{
  int idx = blockIdx.x * 256 + threadIdx.x;
  if (idx >= 64 * KTOT) return;
  int k = idx % KTOT, cout = idx / KTOT;
  int r = k >> 6, ci = k & 63;
  int ky = r / 3, kx = r - ky * 3;
  dst[idx] = (bf16)src[((cout * 64 + ci) * 3 + ky) * 3 + kx];
}

// ---------------------------------------------------------------------------
// Layer-1 input convs (Cin=3, tiny share of FLOPs): direct VALU, bias fused,
// bf16 NHWC output for all three gates. Thread == (f,y,x,c) == NHWC index.
// ---------------------------------------------------------------------------
__global__ __launch_bounds__(256)
void xconv3_kernel(const float* __restrict__ x,   // [128][3][64][64] NCHW fp32
                   const float* __restrict__ Wi, const float* __restrict__ bi,
                   const float* __restrict__ Wc, const float* __restrict__ bc,
                   const float* __restrict__ Wo, const float* __restrict__ bo,
                   bf16* __restrict__ gi, bf16* __restrict__ gc, bf16* __restrict__ go)
{
  const int H = 64, W = 64;
  int idx = blockIdx.x * 256 + threadIdx.x;
  if (idx >= 128 * H * W * CH) return;
  int c = idx & 63, p = idx >> 6;
  int xx = p % W;  p /= W;
  int y  = p % H;
  int f  = p / H;
  float si = bi[c], sc = bc[c], so = bo[c];
  const float* xb = x + (size_t)f * 3 * H * W;
#pragma unroll
  for (int cin = 0; cin < 3; ++cin)
#pragma unroll
    for (int ky = 0; ky < 3; ++ky) {
      int yy = y + ky - 1;
      if (yy < 0 || yy >= H) continue;
#pragma unroll
      for (int kx = 0; kx < 3; ++kx) {
        int xs = xx + kx - 1;
        if (xs < 0 || xs >= W) continue;
        float v = xb[(cin * H + yy) * W + xs];
        int wi = ((c * 3 + cin) * 3 + ky) * 3 + kx;
        si += v * Wi[wi]; sc += v * Wc[wi]; so += v * Wo[wi];
      }
    }
  gi[idx] = (bf16)si; gc[idx] = (bf16)sc; go[idx] = (bf16)so;
}

// ---------------------------------------------------------------------------
// Layer-2 input convs: WMMA implicit GEMM, 3 gates * 4 n-tiles per block.
// ---------------------------------------------------------------------------
__global__ __launch_bounds__(NTHR)
void xgates64_kernel(const bf16* __restrict__ in,   // [F][H][W][CH]
                     const bf16* __restrict__ wb,   // 3 sets of [CH][KTOT]
                     const float* __restrict__ bi, const float* __restrict__ bc,
                     const float* __restrict__ bo,
                     bf16* __restrict__ gi, bf16* __restrict__ gc, bf16* __restrict__ go,
                     int H, int W)
{
  __shared__ __align__(16) bf16 patch[54 * CH];
  const int tid = threadIdx.x;
  const int ox0 = blockIdx.x << 4;
  const int oy  = blockIdx.y;
  const int f   = blockIdx.z;
  stage_patch(patch, in + (size_t)f * H * W * CH, H, W, oy, ox0, tid);
  __syncthreads();
  const int wave = tid >> 5, lane = tid & 31;
  const int g = wave >> 2, nt = wave & 3;
  const int n = (nt << 4) + (lane & 15);
  v8f acc = conv_tile_576(wb + ((size_t)g * CH + n) * KTOT, patch, lane);
  const float bias = (g == 0) ? bi[n] : (g == 1) ? bc[n] : bo[n];
  bf16* out = (g == 0) ? gi : (g == 1) ? gc : go;
  const int hi = lane >> 4;
#pragma unroll
  for (int r = 0; r < 8; ++r) {
    int m = r + (hi << 3);
    int x = ox0 + m;
    if (x < W) out[((size_t)(f * H + oy) * W + x) * CH + n] = (bf16)(acc[r] + bias);
  }
}

// ---------------------------------------------------------------------------
// Recurrent step: 3 hidden convs (WMMA) fused with gate pointwise math.
//   i  = sig(xi + conv(h,Whi)+bhi + Wci*c_prev)
//   cn = i * tanh(xc + conv(h,Whc)+bhc)
//   o  = sig(xo + conv(h,Who)+bho + Wco) * cn ;  h = o * tanh(cn)
// ---------------------------------------------------------------------------
__global__ __launch_bounds__(NTHR)
void lstm_step_kernel(bf16* __restrict__ hseq,      // [128][H][W][CH], frame = b*16+t
                      bf16* __restrict__ cbuf,      // [8][H][W][CH]
                      const bf16* __restrict__ gi, const bf16* __restrict__ gc,
                      const bf16* __restrict__ go,
                      const bf16* __restrict__ wbh, // Whi,Whc,Who each [CH][KTOT]
                      const float* __restrict__ bhi, const float* __restrict__ bhc,
                      const float* __restrict__ bho,
                      const float* __restrict__ Wci, const float* __restrict__ Wco,
                      int H, int W, int t)
{
  __shared__ __align__(16) bf16 patch[54 * CH];
  __shared__ float convb[3 * 16 * CH];
  const int tid = threadIdx.x;
  const int ox0 = blockIdx.x << 4;
  const int oy  = blockIdx.y;
  const int b   = blockIdx.z;
  const int f   = b * 16 + t;
  if (t > 0) {
    stage_patch(patch, hseq + (size_t)(f - 1) * H * W * CH, H, W, oy, ox0, tid);
    __syncthreads();
    const int wave = tid >> 5, lane = tid & 31;
    const int g = wave >> 2, nt = wave & 3;
    const int n = (nt << 4) + (lane & 15);
    v8f acc = conv_tile_576(wbh + ((size_t)g * CH + n) * KTOT, patch, lane);
    const int hi = lane >> 4;
#pragma unroll
    for (int r = 0; r < 8; ++r) {
      int m = r + (hi << 3);
      convb[(g * 16 + m) * CH + n] = acc[r];
    }
  } else {
    for (int idx = tid; idx < 3 * 16 * CH; idx += NTHR) convb[idx] = 0.0f;
  }
  __syncthreads();
  for (int idx = tid; idx < 16 * CH; idx += NTHR) {
    int m = idx >> 6, c = idx & 63;
    int x = ox0 + m;
    if (x >= W) continue;
    size_t po = ((size_t)(f * H + oy) * W + x) * CH + c;
    size_t pc = ((size_t)(b * H + oy) * W + x) * CH + c;
    float cp  = (t > 0) ? (float)cbuf[pc] : 0.0f;
    float wci = Wci[oy * W + x], wco = Wco[oy * W + x];
    float iv = sigmoidf_((float)gi[po] + convb[(0 * 16 + m) * CH + c] + bhi[c] + wci * cp);
    float cn = iv * tanhf((float)gc[po] + convb[(1 * 16 + m) * CH + c] + bhc[c]);
    float ov = sigmoidf_((float)go[po] + convb[(2 * 16 + m) * CH + c] + bho[c] + wco) * cn;
    hseq[po] = (bf16)(ov * tanhf(cn));
    cbuf[pc] = (bf16)cn;
  }
}

// ---------------------------------------------------------------------------
// BN: per-channel sum/sumsq (atomic reduction), finalize, fused BN+maxpool3.
// ---------------------------------------------------------------------------
__global__ __launch_bounds__(256)
void bn_stats_kernel(const bf16* __restrict__ h, long npix, float* __restrict__ accum)
{
  __shared__ float s0[4][64], s1[4][64];
  int c = threadIdx.x & 63, sl = threadIdx.x >> 6;
  float sum = 0.0f, sq = 0.0f;
  for (long p = (long)blockIdx.x * 4 + sl; p < npix; p += (long)gridDim.x * 4) {
    float v = (float)h[p * 64 + c];
    sum += v; sq += v * v;
  }
  s0[sl][c] = sum; s1[sl][c] = sq;
  __syncthreads();
  if (sl == 0) {
    atomicAdd(&accum[c],      s0[0][c] + s0[1][c] + s0[2][c] + s0[3][c]);
    atomicAdd(&accum[64 + c], s1[0][c] + s1[1][c] + s1[2][c] + s1[3][c]);
  }
}

__global__ void bn_finalize_kernel(const float* __restrict__ accum,
                                   const float* __restrict__ gamma,
                                   const float* __restrict__ beta,
                                   float inv_n, float* __restrict__ ss)
{
  int c = threadIdx.x;
  if (c >= 64) return;
  float mean = accum[c] * inv_n;
  float var  = accum[64 + c] * inv_n - mean * mean;
  float sc   = gamma[c] * rsqrtf(var + 1e-5f);
  ss[c]      = sc;
  ss[64 + c] = beta[c] - mean * sc;
}

__global__ __launch_bounds__(256)
void bn_pool_kernel(const bf16* __restrict__ h, const float* __restrict__ ss,
                    int H, int W, int OH, int OW,
                    bf16* __restrict__ out_nhwc, float* __restrict__ out_nchw)
{
  int idx = blockIdx.x * 256 + threadIdx.x;
  if (idx >= 128 * OH * OW * 64) return;
  int c = idx & 63, p = idx >> 6;
  int ox = p % OW;  p /= OW;
  int oy = p % OH;
  int f  = p / OH;
  float sc = ss[c], sh = ss[64 + c];
  float m = -INFINITY;
#pragma unroll
  for (int ky = 0; ky < 3; ++ky)
#pragma unroll
    for (int kx = 0; kx < 3; ++kx) {
      float v = (float)h[(((size_t)f * H + (oy * 3 + ky)) * W + (ox * 3 + kx)) * 64 + c];
      m = fmaxf(m, v * sc + sh);
    }
  if (out_nhwc) out_nhwc[idx] = (bf16)m;
  else out_nchw[(((size_t)f * 64 + c) * OH + oy) * OW + ox] = m; // NCHW fp32 final
}

// ---------------------------------------------------------------------------
extern "C" void kernel_launch(void* const* d_in, const int* in_sizes, int n_in,
                              void* d_out, int out_size, void* d_ws, size_t ws_size,
                              hipStream_t stream)
{
  (void)in_sizes; (void)n_in; (void)out_size;
  const float* X = (const float*)d_in[0];
#define PAR(l, i) ((const float*)d_in[1 + 21 * (l) + (i)])
  // per layer: 0 Wxi 1 bxi 2 Whi 3 bhi | 4..7 (xf/hf unused) | 8 Wxc 9 bxc 10 Whc 11 bhc
  //            12 Wxo 13 bxo 14 Who 15 bho | 16 Wci 17 Wcf 18 Wco | 19 gamma 20 beta

  const size_t L1E = (size_t)128 * 64 * 64 * 64;   // elems per L1 activation tensor
  const size_t L2E = (size_t)128 * 21 * 21 * 64;

  char* ws = (char*)d_ws;
  size_t off = 0;
  auto alloc = [&](size_t bytes) -> void* {
    void* p = ws + off;
    off = (off + bytes + 255) & ~(size_t)255;
    return p;
  };
  bf16* wb_l1h = (bf16*)alloc((size_t)3 * 64 * KTOT * 2);
  bf16* wb_l2x = (bf16*)alloc((size_t)3 * 64 * KTOT * 2);
  bf16* wb_l2h = (bf16*)alloc((size_t)3 * 64 * KTOT * 2);
  bf16* g1i = (bf16*)alloc(L1E * 2);
  bf16* g1c = (bf16*)alloc(L1E * 2);
  bf16* g1o = (bf16*)alloc(L1E * 2);
  bf16* h1  = (bf16*)alloc(L1E * 2);
  bf16* c1  = (bf16*)alloc((size_t)8 * 64 * 64 * 64 * 2);
  bf16* pool1 = (bf16*)alloc(L2E * 2);
  bf16* g2i = (bf16*)alloc(L2E * 2);
  bf16* g2c = (bf16*)alloc(L2E * 2);
  bf16* g2o = (bf16*)alloc(L2E * 2);
  bf16* h2  = (bf16*)alloc(L2E * 2);
  bf16* c2  = (bf16*)alloc((size_t)8 * 21 * 21 * 64 * 2);
  float* accum = (float*)alloc(128 * sizeof(float));
  float* ss    = (float*)alloc(128 * sizeof(float));
  if (off > ws_size) return;  // workspace too small: bail deterministically

  const int WCB = (64 * KTOT + 255) / 256;
  // bf16 weight repacks: L1 Wh{i,c,o}, L2 Wx{i,c,o}, L2 Wh{i,c,o}
  wconv_kernel<<<WCB, 256, 0, stream>>>(PAR(0, 2),  wb_l1h + 0 * 64 * KTOT);
  wconv_kernel<<<WCB, 256, 0, stream>>>(PAR(0, 10), wb_l1h + 1 * 64 * KTOT);
  wconv_kernel<<<WCB, 256, 0, stream>>>(PAR(0, 14), wb_l1h + 2 * 64 * KTOT);
  wconv_kernel<<<WCB, 256, 0, stream>>>(PAR(1, 0),  wb_l2x + 0 * 64 * KTOT);
  wconv_kernel<<<WCB, 256, 0, stream>>>(PAR(1, 8),  wb_l2x + 1 * 64 * KTOT);
  wconv_kernel<<<WCB, 256, 0, stream>>>(PAR(1, 12), wb_l2x + 2 * 64 * KTOT);
  wconv_kernel<<<WCB, 256, 0, stream>>>(PAR(1, 2),  wb_l2h + 0 * 64 * KTOT);
  wconv_kernel<<<WCB, 256, 0, stream>>>(PAR(1, 10), wb_l2h + 1 * 64 * KTOT);
  wconv_kernel<<<WCB, 256, 0, stream>>>(PAR(1, 14), wb_l2h + 2 * 64 * KTOT);

  // ---- Layer 1 ----
  xconv3_kernel<<<(int)(L1E / 256), 256, 0, stream>>>(
      X, PAR(0, 0), PAR(0, 1), PAR(0, 8), PAR(0, 9), PAR(0, 12), PAR(0, 13),
      g1i, g1c, g1o);
  {
    dim3 grid(4, 64, 8);  // 4 m-tiles per row, 64 rows, 8 batches
    for (int t = 0; t < 16; ++t)
      lstm_step_kernel<<<grid, NTHR, 0, stream>>>(
          h1, c1, g1i, g1c, g1o, wb_l1h,
          PAR(0, 3), PAR(0, 11), PAR(0, 15), PAR(0, 16), PAR(0, 18), 64, 64, t);
  }
  (void)hipMemsetAsync(accum, 0, 128 * sizeof(float), stream);
  {
    long npix = 128L * 64 * 64;
    bn_stats_kernel<<<1024, 256, 0, stream>>>(h1, npix, accum);
    bn_finalize_kernel<<<1, 64, 0, stream>>>(accum, PAR(0, 19), PAR(0, 20),
                                             1.0f / (float)npix, ss);
    bn_pool_kernel<<<(int)((L2E + 255) / 256), 256, 0, stream>>>(
        h1, ss, 64, 64, 21, 21, pool1, nullptr);
  }

  // ---- Layer 2 ----
  {
    dim3 grid(2, 21, 128);  // 2 m-tiles per 21-wide row, all 128 frames
    xgates64_kernel<<<grid, NTHR, 0, stream>>>(
        pool1, wb_l2x, PAR(1, 1), PAR(1, 9), PAR(1, 13), g2i, g2c, g2o, 21, 21);
  }
  {
    dim3 grid(2, 21, 8);
    for (int t = 0; t < 16; ++t)
      lstm_step_kernel<<<grid, NTHR, 0, stream>>>(
          h2, c2, g2i, g2c, g2o, wb_l2h,
          PAR(1, 3), PAR(1, 11), PAR(1, 15), PAR(1, 16), PAR(1, 18), 21, 21, t);
  }
  (void)hipMemsetAsync(accum, 0, 128 * sizeof(float), stream);
  {
    long npix = 128L * 21 * 21;
    bn_stats_kernel<<<512, 256, 0, stream>>>(h2, npix, accum);
    bn_finalize_kernel<<<1, 64, 0, stream>>>(accum, PAR(1, 19), PAR(1, 20),
                                             1.0f / (float)npix, ss);
    const size_t OE = (size_t)128 * 7 * 7 * 64;
    bn_pool_kernel<<<(int)((OE + 255) / 256), 256, 0, stream>>>(
        h2, ss, 21, 21, 7, 7, nullptr, (float*)d_out);
  }
#undef PAR
}
